// CloseSerializedAttn_26551487824259
// MI455X (gfx1250) — matrix operand order repaired and checked by reading, not compiled, not run on hardware
//
#include <hip/hip_runtime.h>
#include <hip/hip_bf16.h>

// ---------------------------------------------------------------------------
// Serialized patch attention for MI455X (gfx1250, wave32, WMMA + TDM).
//   kernel 1: qkv = gather(feat, order) @ W_qkv + b ; q *= SCALE ; store f16
//   kernel 2: per (patch, head) dense softmax attention, WMMA f16->f32,
//             q/k tiles staged via TENSOR_LOAD_TO_LDS (TDM, padded rows)
//   kernel 3: out = gather(attn, inverse) @ W_proj + b_proj ; store f32
// Memory-bound (~1.6 GB @ 23.3 TB/s): f16 intermediates halve qkv traffic;
// all LDS->fragment reads are two ds_load_b128 each (B tiles stored N-major).
// ---------------------------------------------------------------------------

typedef __attribute__((ext_vector_type(16))) _Float16 v16h;
typedef __attribute__((ext_vector_type(8)))  _Float16 v8h;
typedef __attribute__((ext_vector_type(4)))  _Float16 v4h;
typedef __attribute__((ext_vector_type(8)))  float    v8f;
typedef __attribute__((ext_vector_type(4)))  unsigned int u32x4;
typedef __attribute__((ext_vector_type(8)))  int          i32x8;
typedef __attribute__((ext_vector_type(4)))  int          i32x4;

#define NPTS   262144
#define CCH    256
#define NHEAD  8
#define KPATCH 128
#define DHEAD  32
#define QKVC   768
#define NPATCH (NPTS / KPATCH)

__constant__ const float kScaleQ = 0.17677669529663687f; // 1/sqrt(32)

// ---------------- WMMA + fragment helpers (layouts per CDNA5 ISA 7.12.2) ---

__device__ __forceinline__ v8f wmma_f16(v16h a, v16h b, v8f c) {
  return __builtin_amdgcn_wmma_f32_16x16x32_f16(
      /*neg_a=*/false, a, /*neg_b=*/false, b,
      /*c_mod=*/(short)0, c, /*reuse_a=*/false, /*reuse_b=*/false);
}

__device__ __forceinline__ v16h cat8(v8h lo, v8h hi) {
  return __builtin_shufflevector(lo, hi, 0, 1, 2, 3, 4, 5, 6, 7,
                                 8, 9, 10, 11, 12, 13, 14, 15);
}

// A-matrix 16x32 f16, row-major source, stride ld halfs (ld*2 % 16 == 0).
// lane&15 = row M; lane>=16 selects K ranges {8..15, 24..31}.
// Two 16B ds_load_b128 per fragment.
__device__ __forceinline__ v16h frag_a(const _Float16* base, int ld, int lane) {
  const _Float16* p = base + (lane & 15) * ld + 8 * (lane >> 4);
  return cat8(*(const v8h*)p, *(const v8h*)(p + 16));
}

// B-matrix 32x16 f16 from N x K storage (B[k][n] = src[n][k]), stride ld.
// lane&15 = col N; lane>=16 selects K ranges {8..15, 24..31} (same as A).
__device__ __forceinline__ v16h frag_b_nk(const _Float16* base, int ld, int lane) {
  const _Float16* p = base + (lane & 15) * ld + 8 * (lane >> 4);
  return cat8(*(const v8h*)p, *(const v8h*)(p + 16));
}

// xor-swizzle reductions confined to each 16-lane half (masks 1,2,4,8),
// matching the C/D layout where one S row spans 16 lanes of one half.
template <int MASK>
__device__ __forceinline__ float swz_xor(float v) {
  return __int_as_float(
      __builtin_amdgcn_ds_swizzle(__float_as_int(v), (0x1f | (MASK << 10))));
}
__device__ __forceinline__ float red_max16(float v) {
  v = fmaxf(v, swz_xor<1>(v)); v = fmaxf(v, swz_xor<2>(v));
  v = fmaxf(v, swz_xor<4>(v)); v = fmaxf(v, swz_xor<8>(v));
  return v;
}
__device__ __forceinline__ float red_sum16(float v) {
  v += swz_xor<1>(v); v += swz_xor<2>(v);
  v += swz_xor<4>(v); v += swz_xor<8>(v);
  return v;
}

// ---------------- Tensor Data Mover: 2-D f16 tile, global -> padded LDS ----
// Loads tile_d1 rows x tile_d0 halfs (row stride = stride_elems halfs) into
// LDS at lds_off, inserting 4 DWORDs of pad after every 16 DWORDs (i.e. LDS
// row stride = 40 halfs for a 32-half row). D# layout per ISA 8.3/8.4.
__device__ __forceinline__ void tdm_load_tile_f16(unsigned lds_off,
                                                  const void* gaddr,
                                                  unsigned tile_d0,
                                                  unsigned tile_d1,
                                                  unsigned stride_elems) {
  const unsigned long long ga = (unsigned long long)(uintptr_t)gaddr;
  u32x4 g0;
  g0[0] = 1u;                                    // count=1, user descriptor
  g0[1] = lds_off;                               // lds_addr (bytes)
  g0[2] = (unsigned)(ga & 0xFFFFFFFFu);          // global_addr[31:0]
  g0[3] = (unsigned)((ga >> 32) & 0x01FFFFFFu)   // global_addr[56:32]
        | (2u << 30);                            // type=2 ("image")
  i32x8 g1;
  g1[0] = (1 << 16)                              // data_size = 2 bytes
        | (1 << 20)                              // pad_enable
        | (3 << 22)                              // pad_interval = 16 DWORDs
        | (3 << 25);                             // pad_amount   = 4 DWORDs
  g1[1] = (int)((tile_d0 & 0xFFFFu) << 16);      // tensor_dim0[15:0] @63:48
  g1[2] = (int)((tile_d1 & 0xFFFFu) << 16);      // dim0[31:16]=0 | dim1[15:0]
  g1[3] = (int)((tile_d0 & 0xFFFFu) << 16);      // dim1[31:16]=0 | tile_dim0
  g1[4] = (int)(tile_d1 & 0xFFFFu);              // tile_dim1 | tile_dim2=0
  g1[5] = (int)stride_elems;                     // tensor_dim0_stride[31:0]
  g1[6] = 0;                                     // stride[47:32] | dim1_stride
  g1[7] = 0;
  const i32x4 gz = {0, 0, 0, 0};
#if __clang_major__ >= 23
  const i32x8 gz8 = {0, 0, 0, 0, 0, 0, 0, 0};
  __builtin_amdgcn_tensor_load_to_lds(g0, g1, gz, gz, gz8, 0);
#else
  __builtin_amdgcn_tensor_load_to_lds(g0, g1, gz, gz, 0);
#endif
}

// ---------------- Kernel 1: QKV GEMM + order-gather + q-scale --------------
// Tile: 128 rows x 128 cols, 256 threads = 8 waves, each wave a 32x64 block.
__global__ __launch_bounds__(256)
void qkv_gemm_kernel(const float* __restrict__ feat,
                     const float* __restrict__ Wqkv,
                     const float* __restrict__ bqkv,
                     const int*   __restrict__ order,
                     _Float16*    __restrict__ qkv_s) {
  __shared__ _Float16 As[128][40];    // 128x32 A tile, row-major (padded)
  __shared__ _Float16 BsT[128][40];   // 32x128 B tile stored N-major (padded)
  __shared__ int rowIdx[128];

  const int tid  = threadIdx.x;
  const int lane = tid & 31;
  const int wave = tid >> 5;
  const int n0   = blockIdx.x * 128;      // 0..5  (768 cols)
  const int m0   = blockIdx.y * 128;      // 0..2047

  if (tid < 128) rowIdx[tid] = order[m0 + tid];

  const int wm = (wave >> 1) * 32;        // wave row offset
  const int wn = (wave & 1) * 64;         // wave col offset

  const v8f vzero = {0.f, 0.f, 0.f, 0.f, 0.f, 0.f, 0.f, 0.f};
  v8f acc[2][4];
#pragma unroll
  for (int i = 0; i < 2; ++i)
#pragma unroll
    for (int j = 0; j < 4; ++j) acc[i][j] = vzero;

  for (int kt = 0; kt < CCH; kt += 32) {
    __syncthreads();
    { // A: 128x32 f32 gathered rows -> f16 LDS (packed v4h stores)
      const int row = tid >> 1;
      const int c0  = (tid & 1) * 16;
      const float* src = feat + (size_t)rowIdx[row] * CCH + kt + c0;
#pragma unroll
      for (int i = 0; i < 16; i += 4) {
        float4 f = *(const float4*)(src + i);
        v4h h = {(_Float16)f.x, (_Float16)f.y, (_Float16)f.z, (_Float16)f.w};
        *(v4h*)&As[row][c0 + i] = h;
      }
    }
    { // B: 32x128 f32 -> f16 LDS, transposed to N-major
      const int kr = tid >> 3;
      const int c0 = (tid & 7) * 16;
      const float* src = Wqkv + (size_t)(kt + kr) * QKVC + n0 + c0;
#pragma unroll
      for (int i = 0; i < 16; i += 4) {
        float4 f = *(const float4*)(src + i);
        BsT[c0 + i + 0][kr] = (_Float16)f.x;
        BsT[c0 + i + 1][kr] = (_Float16)f.y;
        BsT[c0 + i + 2][kr] = (_Float16)f.z;
        BsT[c0 + i + 3][kr] = (_Float16)f.w;
      }
    }
    __syncthreads();

    const v16h a0 = frag_a(&As[wm][0],      40, lane);
    const v16h a1 = frag_a(&As[wm + 16][0], 40, lane);
#pragma unroll
    for (int j = 0; j < 4; ++j) {
      const v16h b = frag_b_nk(&BsT[wn + j * 16][0], 40, lane);
      acc[0][j] = wmma_f16(a0, b, acc[0][j]);
      acc[1][j] = wmma_f16(a1, b, acc[1][j]);
    }
  }

  // epilogue: + bias, scale q columns, store f16 in serialized row order
  const int cl = lane & 15;
  const int hi = lane >> 4;
#pragma unroll
  for (int i = 0; i < 2; ++i) {
#pragma unroll
    for (int j = 0; j < 4; ++j) {
      const int col  = n0 + wn + j * 16 + cl;
      const float bi = bqkv[col];
      const float sc = (col < CCH) ? kScaleQ : 1.0f;  // q component only
#pragma unroll
      for (int r = 0; r < 8; ++r) {
        const int row = m0 + wm + i * 16 + r + 8 * hi;
        qkv_s[(size_t)row * QKVC + col] = (_Float16)((acc[i][j][r] + bi) * sc);
      }
    }
  }
}

// ---------------- Kernel 2: per-(patch, head) dense attention --------------
// One workgroup = one (patch, head). 8 waves; wave w owns S rows [16w,16w+16).
// q/k tiles DMA'd by the Tensor Data Mover; v staged transposed by VALU.
__global__ __launch_bounds__(256)
void attn_kernel(const _Float16* __restrict__ qkv_s,
                 _Float16* __restrict__ attn_s) {
  __shared__ _Float16 qs[KPATCH][40];    // q, row-major, TDM-padded rows
  __shared__ _Float16 ks[KPATCH][40];    // k, row-major, TDM-padded rows
  __shared__ _Float16 vT[DHEAD][136];    // v transposed: [d][token]
  __shared__ _Float16 ps[KPATCH][136];   // softmax probs, f16, row-major

  const int tid  = threadIdx.x;
  const int lane = tid & 31;
  const int wave = tid >> 5;
  const int p    = blockIdx.x >> 3;
  const int h    = blockIdx.x & 7;

  const _Float16* gq =
      qkv_s + (size_t)(p * KPATCH) * QKVC + h * DHEAD;            // q tile
  // TDM: one DMA per issuing wave (EXEC ignored); wave 0 -> q, wave 1 -> k.
  if (wave == 0) {
    tdm_load_tile_f16((unsigned)(uintptr_t)&qs[0][0], gq, DHEAD, KPATCH, QKVC);
  } else if (wave == 1) {
    tdm_load_tile_f16((unsigned)(uintptr_t)&ks[0][0], gq + CCH, DHEAD, KPATCH,
                      QKVC);
  }

  { // v: stage transposed [d][token] for contiguous B-fragment reads
    const int tok  = tid >> 1;
    const int part = (tid & 1) * 16;
    const _Float16* vsrc =
        qkv_s + (size_t)(p * KPATCH + tok) * QKVC + 2 * CCH + h * DHEAD + part;
    v8h a0 = *(const v8h*)(vsrc);
    v8h a1 = *(const v8h*)(vsrc + 8);
#pragma unroll
    for (int i = 0; i < 8; ++i) {
      vT[part + i][tok]     = a0[i];
      vT[part + 8 + i][tok] = a1[i];
    }
  }
  __builtin_amdgcn_s_wait_tensorcnt(0);
  __syncthreads();

  // S = q * k^T : one A frag (16 rows, full d=32), 8 column tiles
  const v8f vzero = {0.f, 0.f, 0.f, 0.f, 0.f, 0.f, 0.f, 0.f};
  v8f s[8];
  {
    const v16h a = frag_a(&qs[16 * wave][0], 40, lane);
#pragma unroll
    for (int nt = 0; nt < 8; ++nt) {
      const v16h b = frag_b_nk(&ks[nt * 16][0], 40, lane);  // B[k=d][n=token]
      s[nt] = wmma_f16(a, b, vzero);
    }
  }

  // row softmax (row = 16*wave + r + 8*hi spans 16 lanes of this half)
#pragma unroll
  for (int r = 0; r < 8; ++r) {
    float m = s[0][r];
#pragma unroll
    for (int nt = 1; nt < 8; ++nt) m = fmaxf(m, s[nt][r]);
    m = red_max16(m);
    float sum = 0.f;
#pragma unroll
    for (int nt = 0; nt < 8; ++nt) {
      const float e = __expf(s[nt][r] - m);
      s[nt][r] = e;
      sum += e;
    }
    sum = red_sum16(sum);
    const float inv = 1.0f / sum;
#pragma unroll
    for (int nt = 0; nt < 8; ++nt) s[nt][r] *= inv;
  }

  // spill normalized P to LDS in f16 for re-fragmenting as A
  {
    const int cl = lane & 15;
    const int hi = lane >> 4;
#pragma unroll
    for (int nt = 0; nt < 8; ++nt)
#pragma unroll
      for (int r = 0; r < 8; ++r)
        ps[16 * wave + r + 8 * hi][nt * 16 + cl] = (_Float16)s[nt][r];
  }
  __syncthreads();

  // O = P * V : K=128 in 4 chunks of 32, two 16-wide N halves (d=32)
  v8f o0 = vzero, o1 = vzero;
#pragma unroll
  for (int kc = 0; kc < 4; ++kc) {
    const v16h a  = frag_a(&ps[16 * wave][kc * 32], 136, lane);
    const v16h b0 = frag_b_nk(&vT[0][kc * 32],  136, lane);   // n = d 0..15
    const v16h b1 = frag_b_nk(&vT[16][kc * 32], 136, lane);   // n = d 16..31
    o0 = wmma_f16(a, b0, o0);
    o1 = wmma_f16(a, b1, o1);
  }

  { // store O (f16, serialized order) into [N, 256]
    const int cl = lane & 15;
    const int hi = lane >> 4;
#pragma unroll
    for (int r = 0; r < 8; ++r) {
      const int tokg = p * KPATCH + 16 * wave + r + 8 * hi;
      _Float16* dst = attn_s + (size_t)tokg * CCH + h * DHEAD + cl;
      dst[0]  = (_Float16)o0[r];
      dst[16] = (_Float16)o1[r];
    }
  }
}

// ---------------- Kernel 3: projection GEMM + inverse-gather ---------------
__global__ __launch_bounds__(256)
void proj_gemm_kernel(const _Float16* __restrict__ attn_s,
                      const float* __restrict__ Wproj,
                      const float* __restrict__ bproj,
                      const int*   __restrict__ inverse,
                      float* __restrict__ out) {
  __shared__ _Float16 As[128][40];
  __shared__ _Float16 BsT[128][40];   // B tile stored N-major
  __shared__ int rowIdx[128];

  const int tid  = threadIdx.x;
  const int lane = tid & 31;
  const int wave = tid >> 5;
  const int n0   = blockIdx.x * 128;   // 0..1 (256 cols)
  const int m0   = blockIdx.y * 128;

  if (tid < 128) rowIdx[tid] = inverse[m0 + tid];

  const int wm = (wave >> 1) * 32;
  const int wn = (wave & 1) * 64;

  const v8f vzero = {0.f, 0.f, 0.f, 0.f, 0.f, 0.f, 0.f, 0.f};
  v8f acc[2][4];
#pragma unroll
  for (int i = 0; i < 2; ++i)
#pragma unroll
    for (int j = 0; j < 4; ++j) acc[i][j] = vzero;

  for (int kt = 0; kt < CCH; kt += 32) {
    __syncthreads();
    { // A: f16 rows gathered via inverse (pure 16B copies)
      const int row  = tid >> 1;
      const int part = (tid & 1) * 16;
      const float4* src =
          (const float4*)(attn_s + (size_t)rowIdx[row] * CCH + kt + part);
      *(float4*)&As[row][part]     = src[0];
      *(float4*)&As[row][part + 8] = src[1];
    }
    { // B: 32x128 f32 -> f16 LDS, transposed to N-major
      const int kr = tid >> 3;
      const int c0 = (tid & 7) * 16;
      const float* src = Wproj + (size_t)(kt + kr) * CCH + n0 + c0;
#pragma unroll
      for (int i = 0; i < 16; i += 4) {
        float4 f = *(const float4*)(src + i);
        BsT[c0 + i + 0][kr] = (_Float16)f.x;
        BsT[c0 + i + 1][kr] = (_Float16)f.y;
        BsT[c0 + i + 2][kr] = (_Float16)f.z;
        BsT[c0 + i + 3][kr] = (_Float16)f.w;
      }
    }
    __syncthreads();

    const v16h a0 = frag_a(&As[wm][0],      40, lane);
    const v16h a1 = frag_a(&As[wm + 16][0], 40, lane);
#pragma unroll
    for (int j = 0; j < 4; ++j) {
      const v16h b = frag_b_nk(&BsT[wn + j * 16][0], 40, lane);
      acc[0][j] = wmma_f16(a0, b, acc[0][j]);
      acc[1][j] = wmma_f16(a1, b, acc[1][j]);
    }
  }

  const int cl = lane & 15;
  const int hi = lane >> 4;
#pragma unroll
  for (int i = 0; i < 2; ++i) {
#pragma unroll
    for (int j = 0; j < 4; ++j) {
      const int col  = n0 + wn + j * 16 + cl;
      const float bi = bproj[col];
#pragma unroll
      for (int r = 0; r < 8; ++r) {
        const int row = m0 + wm + i * 16 + r + 8 * hi;
        out[(size_t)row * CCH + col] = acc[i][j][r] + bi;
      }
    }
  }
}

// ---------------- host entry ----------------------------------------------
extern "C" void kernel_launch(void* const* d_in, const int* in_sizes, int n_in,
                              void* d_out, int out_size, void* d_ws,
                              size_t ws_size, hipStream_t stream) {
  const float* feat    = (const float*)d_in[0];
  const float* Wqkv    = (const float*)d_in[1];
  const float* bqkv    = (const float*)d_in[2];
  const float* Wproj   = (const float*)d_in[3];
  const float* bproj   = (const float*)d_in[4];
  const int*   order   = (const int*)d_in[5];
  const int*   inverse = (const int*)d_in[6];
  float* out = (float*)d_out;

  // workspace: qkv f16 [N,768] then attn f16 [N,256]  (~537 MB total)
  _Float16* qkv_s  = (_Float16*)d_ws;
  _Float16* attn_s = qkv_s + (size_t)NPTS * QKVC;

  qkv_gemm_kernel<<<dim3(QKVC / 128, NPTS / 128), 256, 0, stream>>>(
      feat, Wqkv, bqkv, order, qkv_s);
  attn_kernel<<<dim3(NPATCH * NHEAD), 256, 0, stream>>>(qkv_s, attn_s);
  proj_gemm_kernel<<<dim3(CCH / 128, NPTS / 128), 256, 0, stream>>>(
      attn_s, Wproj, bproj, inverse, out);
}
